// PD_Block_37684043055283
// MI455X (gfx1250) — matrix operand order repaired and checked by reading, not compiled, not run on hardware
//
#include <hip/hip_runtime.h>
#include <hip/hip_bf16.h>
#include <math.h>

typedef __attribute__((ext_vector_type(16))) __bf16 bf16x16;
typedef __attribute__((ext_vector_type(8)))  float  f32x8;

union Frag { uint4 q[2]; bf16x16 v; };

__device__ __forceinline__ float wredsum(float v) {
    v += __shfl_xor(v, 16);
    v += __shfl_xor(v, 8);
    v += __shfl_xor(v, 4);
    v += __shfl_xor(v, 2);
    v += __shfl_xor(v, 1);
    return v;
}

__device__ __forceinline__ float sigmoidf_(float z) { return 1.0f / (1.0f + expf(-z)); }
__device__ __forceinline__ float siluf_(float z)    { return z / (1.0f + expf(-z)); }

// ---------------------------------------------------------------------------
// Weight prepack: f32 [co][ci][kh][kw] -> bf16 [tap][co][ci]
// ---------------------------------------------------------------------------
__global__ void prepack_kernel(const float* __restrict__ w,
                               __hip_bfloat16* __restrict__ o, int KK) {
    int idx = blockIdx.x * blockDim.x + threadIdx.x;
    int total = 65536 * KK;
    if (idx >= total) return;
    int t  = idx >> 16;
    int co = (idx >> 8) & 255;
    int ci = idx & 255;
    o[idx] = __float2bfloat16(w[(size_t)((co << 8) + ci) * KK + t]);
}

// ---------------------------------------------------------------------------
// GroupNorm: two-stage deterministic reduction over contiguous slices
// ---------------------------------------------------------------------------
__global__ __launch_bounds__(256) void gn_partial_kernel(
    const float* __restrict__ src, float2* __restrict__ part, long groupLen) {
    int bg = blockIdx.x, pi = blockIdx.y, P = gridDim.y;
    const float* s = src + (size_t)bg * groupLen;
    size_t chunk = (size_t)groupLen / P;
    size_t base = (size_t)pi * chunk;
    float su = 0.f, sq = 0.f;
    for (size_t i = base + (size_t)threadIdx.x * 4; i < base + chunk; i += 1024) {
        float4 v = *reinterpret_cast<const float4*>(s + i);
        su += v.x + v.y + v.z + v.w;
        sq += v.x * v.x + v.y * v.y + v.z * v.z + v.w * v.w;
    }
    __shared__ float rs[256], rq[256];
    int tid = threadIdx.x;
    rs[tid] = su; rq[tid] = sq;
    __syncthreads();
    for (int st = 128; st > 0; st >>= 1) {
        if (tid < st) { rs[tid] += rs[tid + st]; rq[tid] += rq[tid + st]; }
        __syncthreads();
    }
    if (tid == 0) part[(size_t)bg * P + pi] = make_float2(rs[0], rq[0]);
}

__global__ __launch_bounds__(256) void gn_final_kernel(
    const float2* __restrict__ part, float2* __restrict__ stats, float invLen) {
    int bg = blockIdx.x, tid = threadIdx.x;
    float2 p = part[(size_t)bg * 256 + tid];
    __shared__ float rs[256], rq[256];
    rs[tid] = p.x; rq[tid] = p.y;
    __syncthreads();
    for (int st = 128; st > 0; st >>= 1) {
        if (tid < st) { rs[tid] += rs[tid + st]; rq[tid] += rq[tid + st]; }
        __syncthreads();
    }
    if (tid == 0) {
        float mean = rs[0] * invLen;
        float var  = rq[0] * invLen - mean * mean;
        stats[bg] = make_float2(mean, rsqrtf(var + 1e-5f));
    }
}

// NCHW f32 -> NHWC {f32|bf16}, affine + optional SiLU, via LDS transpose tile
__global__ __launch_bounds__(256) void gn_apply_t_kernel(
    const float* __restrict__ in, void* __restrict__ outp,
    const float* __restrict__ gw, const float* __restrict__ gb,
    const float2* __restrict__ stats, int G, int doSilu, int outBf16) {
    __shared__ float tile[32][65];
    int c0 = blockIdx.x * 32;
    int p0 = blockIdx.y * 64;
    int tid = threadIdx.x;
    int col = tid & 63;
    int rgrp = tid >> 6;                 // 0..3
    int pglob = p0 + col;
    int b = pglob >> 14;
    int psp = pglob & 16383;
#pragma unroll
    for (int k = 0; k < 8; k++) {
        int cl = rgrp + (k << 2);
        int c = c0 + cl;
        int g = (c * G) >> 8;
        float2 st = stats[b * G + g];
        float v = in[((size_t)(b * 256 + c) << 14) + psp];
        tile[cl][col] = (v - st.x) * st.y * gw[c] + gb[c];
    }
    __syncthreads();
    int cw = tid & 31;
    int prow = tid >> 5;                 // 0..7
#pragma unroll
    for (int k = 0; k < 8; k++) {
        int p2 = prow + (k << 3);
        float v = tile[cw][p2];
        if (doSilu) v = siluf_(v);
        size_t oidx = (((size_t)(p0 + p2)) << 8) + c0 + cw;
        if (outBf16) ((__hip_bfloat16*)outp)[oidx] = __float2bfloat16(v);
        else         ((float*)outp)[oidx] = v;
    }
}

// NHWC f32 -> NHWC bf16, groups=1 affine (no layout change)
__global__ __launch_bounds__(256) void gn_apply_nhwc_kernel(
    const float* __restrict__ in, __hip_bfloat16* __restrict__ out,
    const float* __restrict__ gw, const float* __restrict__ gb,
    const float2* __restrict__ stats) {
    size_t i = (size_t)blockIdx.x * 256 + threadIdx.x;
    int c = (int)(i & 255);
    int b = (int)(i >> 22);
    float2 st = stats[b];
    float v = in[i];
    out[i] = __float2bfloat16((v - st.x) * st.y * gw[c] + gb[c]);
}

// ---------------------------------------------------------------------------
// Cluster attention: 1 workgroup (8 waves) per sub-batch; lane == channel
// ---------------------------------------------------------------------------
__global__ __launch_bounds__(256) void cluster_kernel(
    const float* __restrict__ xn,      // NHWC f32 [2,16384,256]
    float* __restrict__ clout,         // NHWC f32
    float* __restrict__ simws,         // [64][4][4096]
    const float* __restrict__ salpha, const float* __restrict__ sbeta) {
    __shared__ float cfP[8][4][32];
    __shared__ float SP[8][4];
    __shared__ float cfL[4][32];
    __shared__ float cfnL[4][32];
    __shared__ float denseL[4][32];

    int sb = blockIdx.x;
    int bh = sb >> 2, fw = (sb >> 1) & 1, fh = sb & 1;
    int Bidx = bh >> 3, head = bh & 7;
    int chb = head * 32;
    int Wb = fw * 64, Hb = fh * 64;
    int tid = threadIdx.x, lane = tid & 31, g = tid >> 5;
    float alpha = salpha[0], beta = sbeta[0];
    size_t simbase = (size_t)sb * 16384;

    auto xidx = [&](int n) -> size_t {
        int w = n >> 6, h = n & 63;
        int psp = (Wb + w) * 128 + (Hb + h);
        return (((size_t)(Bidx * 16384 + psp)) << 8) + chb + lane;
    };

    // phase 0: centers cf[m][c] = mean over 32x32 pooling block
    float cfa0 = 0.f, cfa1 = 0.f, cfa2 = 0.f, cfa3 = 0.f;
    for (int n = g; n < 4096; n += 8) {
        int w = n >> 6, h = n & 63;
        int m = ((w >> 5) << 1) | (h >> 5);
        float v = xn[xidx(n)];
        cfa0 += (m == 0) ? v : 0.f;
        cfa1 += (m == 1) ? v : 0.f;
        cfa2 += (m == 2) ? v : 0.f;
        cfa3 += (m == 3) ? v : 0.f;
    }
    cfP[g][0][lane] = cfa0; cfP[g][1][lane] = cfa1;
    cfP[g][2][lane] = cfa2; cfP[g][3][lane] = cfa3;
    __syncthreads();
    if (tid < 128) {
        int m = tid >> 5, c = lane;
        float s = 0.f;
        for (int gg = 0; gg < 8; gg++) s += cfP[gg][m][c];
        cfL[m][c] = s * (1.0f / 1024.0f);
    }
    __syncthreads();
    if (g < 4) {
        float v = cfL[g][lane];
        float n2 = wredsum(v * v);
        cfnL[g][lane] = v / fmaxf(sqrtf(n2), 1e-12f);
    }
    __syncthreads();

    // phase 1: sim + weighted accumulation
    float am[4] = {0.f, 0.f, 0.f, 0.f};
    float Sm[4] = {0.f, 0.f, 0.f, 0.f};
    for (int n = g; n < 4096; n += 8) {
        float v = xn[xidx(n)];
        float n2 = wredsum(v * v);
        float xh = v / fmaxf(sqrtf(n2), 1e-12f);
#pragma unroll
        for (int m = 0; m < 4; m++) {
            float d = wredsum(xh * cfnL[m][lane]);
            float sm = sigmoidf_(beta + alpha * d);
            if (lane == m) simws[simbase + m * 4096 + n] = sm;
            am[m] += sm * v;
            Sm[m] += sm;
        }
    }
    __syncthreads();
#pragma unroll
    for (int m = 0; m < 4; m++) cfP[g][m][lane] = am[m];
    if (lane == 0) {
#pragma unroll
        for (int m = 0; m < 4; m++) SP[g][m] = Sm[m];
    }
    __syncthreads();
    if (tid < 128) {
        int m = tid >> 5, c = lane;
        float s = 0.f, St = 0.f;
        for (int gg = 0; gg < 8; gg++) { s += cfP[gg][m][c]; St += SP[gg][m]; }
        denseL[m][c] = (s + cfL[m][c]) / (St + 1.0f);
    }
    __syncthreads();

    // phase 2: dispatch back to points
    for (int n = g; n < 4096; n += 8) {
        float o = 0.f;
#pragma unroll
        for (int m = 0; m < 4; m++) o += denseL[m][lane] * simws[simbase + m * 4096 + n];
        clout[xidx(n)] = o;
    }
}

// ---------------------------------------------------------------------------
// 1x1 proj GEMM (bf16 WMMA) fused: +bias, SiLU, y = x + con_alpha * mix
// Double-buffered LDS staging; loads for chunk k+1 issued before compute k.
// ---------------------------------------------------------------------------
__global__ __launch_bounds__(256) void proj_kernel(
    const __hip_bfloat16* __restrict__ cn,   // NHWC bf16
    const __hip_bfloat16* __restrict__ wp,   // [co][ci] bf16
    const float* __restrict__ bias,
    const float* __restrict__ x,             // NCHW f32
    const float* __restrict__ conA,
    float* __restrict__ y) {                 // NCHW f32
    __shared__ __align__(16) __hip_bfloat16 patch[2][128 * 32];
    const int co_base = blockIdx.x * 64;
    const int row = blockIdx.y;
    const int b = row >> 7, yy = row & 127;
    const int tid = threadIdx.x, lane = tid & 31, wave = tid >> 5;
    const int cw = wave & 3, pw = wave >> 2;
    const int mhalf = (lane >> 4) << 3;
    const int coA = co_base + cw * 16 + (lane & 15);
    const int kbA = (lane < 16) ? 0 : 8;
    const int khB = (lane < 16) ? 0 : 8;

    // staging: 512 uint4 units, 2 per thread
    const int site0 = tid >> 1, part0 = (tid & 1) << 1; // parts {0,1} or {2,3}
    const size_t grow = ((size_t)((b << 14) + (yy << 7) + site0)) << 8;
    uint4 stg[2];

    auto issueLoads = [&](int cin0) {
        stg[0] = *reinterpret_cast<const uint4*>(cn + grow + cin0 + (part0 << 3));
        stg[1] = *reinterpret_cast<const uint4*>(cn + grow + cin0 + ((part0 + 1) << 3));
    };
    auto storeStage = [&](int buf) {
        *reinterpret_cast<uint4*>(&patch[buf][site0 * 32 + (part0 << 3)]) = stg[0];
        *reinterpret_cast<uint4*>(&patch[buf][site0 * 32 + ((part0 + 1) << 3)]) = stg[1];
    };

    f32x8 acc[4];
    {
        const int cob = co_base + cw * 16 + mhalf;
        f32x8 bi;
#pragma unroll
        for (int v = 0; v < 8; v++) bi[v] = bias[cob + v];
        acc[0] = bi; acc[1] = bi; acc[2] = bi; acc[3] = bi;
    }

    issueLoads(0);
    storeStage(0);
    __syncthreads();

    for (int k = 0; k < 8; k++) {
        const int cin0 = k << 5;
        if (k < 7) issueLoads(cin0 + 32);
        const __hip_bfloat16* pbuf = patch[k & 1];

        Frag a;
        const __hip_bfloat16* wb = wp + ((size_t)coA << 8) + cin0 + kbA;
        a.q[0] = *reinterpret_cast<const uint4*>(wb);
        a.q[1] = *reinterpret_cast<const uint4*>(wb + 16);
        Frag bf[4];
#pragma unroll
        for (int t = 0; t < 4; t++) {
            const __hip_bfloat16* pb = pbuf + (((pw << 6) + (t << 4) + (lane & 15)) << 5) + khB;
            bf[t].q[0] = *reinterpret_cast<const uint4*>(pb);
            bf[t].q[1] = *reinterpret_cast<const uint4*>(pb + 16);
        }
#pragma unroll
        for (int t = 0; t < 4; t++)
            acc[t] = __builtin_amdgcn_wmma_f32_16x16x32_bf16(
                false, a.v, false, bf[t].v, (short)0, acc[t], false, false);

        if (k < 7) storeStage((k + 1) & 1);
        __syncthreads();
    }

    float ca = conA[0];
#pragma unroll
    for (int t = 0; t < 4; t++) {
        int xx = (pw << 6) + (t << 4) + (lane & 15);
#pragma unroll
        for (int v = 0; v < 8; v++) {
            int co = co_base + cw * 16 + mhalf + v;
            size_t off = (((size_t)((b << 8) + co)) << 14) + (yy << 7) + xx;
            float val = acc[t][v];
            y[off] = x[off] + ca * siluf_(val);
        }
    }
}

// ---------------------------------------------------------------------------
// 5x5 conv as implicit GEMM (bf16 WMMA), optional fused residual add.
// Double-buffered LDS patch + software-pipelined A-fragment loads.
// ---------------------------------------------------------------------------
__global__ __launch_bounds__(256) void conv5_kernel(
    const __hip_bfloat16* __restrict__ act,  // NHWC bf16 (pre-normed+SiLU)
    const __hip_bfloat16* __restrict__ wp,   // [25][co][ci] bf16
    const float* __restrict__ bias,
    const float* __restrict__ resid,         // NCHW f32 or nullptr
    float* __restrict__ out) {               // NCHW f32
    __shared__ __align__(16) __hip_bfloat16 patch[2][5 * 132 * 32];
    const int co_base = blockIdx.x * 64;
    const int row = blockIdx.y;
    const int b = row >> 7, yy = row & 127;
    const int tid = threadIdx.x, lane = tid & 31, wave = tid >> 5;
    const int cw = wave & 3, pw = wave >> 2;
    const int mhalf = (lane >> 4) << 3;
    const int coA = co_base + cw * 16 + (lane & 15);
    const int kbA = (lane < 16) ? 0 : 8;
    const int khB = (lane < 16) ? 0 : 8;

    // staging: 660 sites * 4 parts = 2640 uint4 units; up to 11 per thread
    uint4 stg[11];
    auto issueLoads = [&](int cin0) {
#pragma unroll
        for (int j = 0; j < 11; j++) {
            int u = tid + (j << 8);
            uint4 val = make_uint4(0u, 0u, 0u, 0u);
            if (u < 2640) {
                int site = u >> 2, part = u & 3;
                int r = site / 132, xl = site - r * 132;
                int ys = yy + r - 2, xs = xl - 2;
                if ((unsigned)ys < 128u && (unsigned)xs < 128u) {
                    size_t gidx = (((size_t)((b << 14) + (ys << 7) + xs)) << 8)
                                  + cin0 + (part << 3);
                    val = *reinterpret_cast<const uint4*>(act + gidx);
                }
            }
            stg[j] = val;
        }
    };
    auto storeStage = [&](int buf) {
#pragma unroll
        for (int j = 0; j < 11; j++) {
            int u = tid + (j << 8);
            if (u < 2640) {
                int site = u >> 2, part = u & 3;
                *reinterpret_cast<uint4*>(&patch[buf][site * 32 + (part << 3)]) = stg[j];
            }
        }
    };
    auto loadA = [&](Frag& a, int tap, int cin0) {
        const __hip_bfloat16* wb =
            wp + ((size_t)tap << 16) + ((size_t)coA << 8) + cin0 + kbA;
        a.q[0] = *reinterpret_cast<const uint4*>(wb);
        a.q[1] = *reinterpret_cast<const uint4*>(wb + 16);
    };

    f32x8 acc[4];
    {
        const int cob = co_base + cw * 16 + mhalf;
        f32x8 bi;
#pragma unroll
        for (int v = 0; v < 8; v++) bi[v] = bias[cob + v];
        acc[0] = bi; acc[1] = bi; acc[2] = bi; acc[3] = bi;
    }

    issueLoads(0);
    storeStage(0);
    __syncthreads();

    for (int k = 0; k < 8; k++) {
        const int cin0 = k << 5;
        if (k < 7) issueLoads(cin0 + 32);       // hide global latency behind WMMAs
        const __hip_bfloat16* pbuf = patch[k & 1];

        Frag aN;
        loadA(aN, 0, cin0);
#pragma unroll
        for (int tap = 0; tap < 25; tap++) {
            const int ky = tap / 5, kx = tap % 5;
            Frag a = aN;
            if (tap < 24) loadA(aN, tap + 1, cin0);  // prefetch next tap's weights
            Frag bf[4];
#pragma unroll
            for (int t = 0; t < 4; t++) {
                int xloc = (pw << 6) + (t << 4) + (lane & 15) + kx;
                const __hip_bfloat16* pb = pbuf + ((ky * 132 + xloc) << 5) + khB;
                bf[t].q[0] = *reinterpret_cast<const uint4*>(pb);
                bf[t].q[1] = *reinterpret_cast<const uint4*>(pb + 16);
            }
#pragma unroll
            for (int t = 0; t < 4; t++)
                acc[t] = __builtin_amdgcn_wmma_f32_16x16x32_bf16(
                    false, a.v, false, bf[t].v, (short)0, acc[t], false, false);
        }

        if (k < 7) storeStage((k + 1) & 1);
        __syncthreads();
    }

#pragma unroll
    for (int t = 0; t < 4; t++) {
        int xx = (pw << 6) + (t << 4) + (lane & 15);
#pragma unroll
        for (int v = 0; v < 8; v++) {
            int co = co_base + cw * 16 + mhalf + v;
            size_t off = (((size_t)((b << 8) + co)) << 14) + (yy << 7) + xx;
            float val = acc[t][v];
            if (resid) val += resid[off];
            out[off] = val;
        }
    }
}

// ---------------------------------------------------------------------------
extern "C" void kernel_launch(void* const* d_in, const int* in_sizes, int n_in,
                              void* d_out, int out_size, void* d_ws, size_t ws_size,
                              hipStream_t stream) {
    (void)in_sizes; (void)n_in; (void)out_size; (void)ws_size;
    const float* x        = (const float*)d_in[0];
    const float* norm1_w  = (const float*)d_in[1];
    const float* norm1_b  = (const float*)d_in[2];
    // d_in[3], d_in[4] (v_w, v_b) are dead: value path contributes nothing
    const float* proj_w   = (const float*)d_in[5];
    const float* proj_b   = (const float*)d_in[6];
    const float* cn_w     = (const float*)d_in[7];
    const float* cn_b     = (const float*)d_in[8];
    const float* sim_alpha= (const float*)d_in[9];
    const float* sim_beta = (const float*)d_in[10];
    const float* con_alpha= (const float*)d_in[11];
    const float* rn1_w    = (const float*)d_in[12];
    const float* rn1_b    = (const float*)d_in[13];
    const float* rc1_w    = (const float*)d_in[14];
    const float* rc1_b    = (const float*)d_in[15];
    const float* rn2_w    = (const float*)d_in[16];
    const float* rn2_b    = (const float*)d_in[17];
    const float* rc2_w    = (const float*)d_in[18];
    const float* rc2_b    = (const float*)d_in[19];

    char* wsb = (char*)d_ws;
    float*          bufA  = (float*)(wsb + 0);                    // 33,554,432 B : xn, then y
    __hip_bfloat16* bufC  = (__hip_bfloat16*)(wsb + 33554432);    // 16,777,216 B : cn / act1 / act2
    __hip_bfloat16* wProj = (__hip_bfloat16*)(wsb + 50331648);    //    131,072 B
    __hip_bfloat16* wRc1  = (__hip_bfloat16*)(wsb + 50462720);    //  3,276,800 B
    __hip_bfloat16* wRc2  = (__hip_bfloat16*)(wsb + 53739520);    //  3,276,800 B
    float*          simws = (float*)(wsb + 57016320);             //  4,194,304 B
    float2*         part  = (float2*)(wsb + 61210624);            //     65,536 B
    float2*         stats = (float2*)(wsb + 61276160);            //        256 B
    float*          outF  = (float*)d_out;                        // clout / t1 scratch + final

    // weight prepack (f32 -> bf16, [tap][co][ci])
    prepack_kernel<<<256, 256, 0, stream>>>(proj_w, wProj, 1);
    prepack_kernel<<<6400, 256, 0, stream>>>(rc1_w, wRc1, 25);
    prepack_kernel<<<6400, 256, 0, stream>>>(rc2_w, wRc2, 25);

    // GN1(x) -> xn (NHWC f32)
    gn_partial_kernel<<<dim3(2, 256), 256, 0, stream>>>(x, part, 4194304L);
    gn_final_kernel<<<2, 256, 0, stream>>>(part, stats, 1.0f / 4194304.0f);
    gn_apply_t_kernel<<<dim3(8, 512), 256, 0, stream>>>(x, bufA, norm1_w, norm1_b, stats, 1, 0, 0);

    // cluster attention -> clout (NHWC f32, stored in d_out as scratch)
    cluster_kernel<<<64, 256, 0, stream>>>(bufA, outF, simws, sim_alpha, sim_beta);

    // GN1(clout) -> cn (NHWC bf16)
    gn_partial_kernel<<<dim3(2, 256), 256, 0, stream>>>(outF, part, 4194304L);
    gn_final_kernel<<<2, 256, 0, stream>>>(part, stats, 1.0f / 4194304.0f);
    gn_apply_nhwc_kernel<<<32768, 256, 0, stream>>>(outF, bufC, cn_w, cn_b, stats);

    // proj 1x1 GEMM + SiLU, y = x + con_alpha*mix (NCHW f32 in bufA)
    proj_kernel<<<dim3(4, 256), 256, 0, stream>>>(bufC, wProj, proj_b, x, con_alpha, bufA);

    // GN16(y) + SiLU -> act1 (NHWC bf16)
    gn_partial_kernel<<<dim3(32, 256), 256, 0, stream>>>(bufA, part, 262144L);
    gn_final_kernel<<<32, 256, 0, stream>>>(part, stats, 1.0f / 262144.0f);
    gn_apply_t_kernel<<<dim3(8, 512), 256, 0, stream>>>(bufA, bufC, rn1_w, rn1_b, stats, 16, 1, 1);

    // conv5x5 #1 -> t1 (NCHW f32, in d_out scratch)
    conv5_kernel<<<dim3(4, 256), 256, 0, stream>>>(bufC, wRc1, rc1_b, nullptr, outF);

    // GN16(t1) + SiLU -> act2 (NHWC bf16)
    gn_partial_kernel<<<dim3(32, 256), 256, 0, stream>>>(outF, part, 262144L);
    gn_final_kernel<<<32, 256, 0, stream>>>(part, stats, 1.0f / 262144.0f);
    gn_apply_t_kernel<<<dim3(8, 512), 256, 0, stream>>>(outF, bufC, rn2_w, rn2_b, stats, 16, 1, 1);

    // conv5x5 #2 + residual y -> final output (NCHW f32)
    conv5_kernel<<<dim3(4, 256), 256, 0, stream>>>(bufC, wRc2, rc2_b, bufA, outF);
}
